// HyperConv2d_86758339379721
// MI455X (gfx1250) — matrix operand order repaired and checked by reading, not compiled, not run on hardware
//
#include <hip/hip_runtime.h>
#include <hip/hip_bf16.h>
#include <math.h>

typedef __attribute__((ext_vector_type(16))) __bf16 v16bf;
typedef __attribute__((ext_vector_type(8)))  float  v8f;
typedef __attribute__((ext_vector_type(4)))  unsigned int v4u;
typedef __attribute__((ext_vector_type(8)))  int v8i;
typedef __attribute__((ext_vector_type(4)))  int v4i;

#define BATCH   16
#define INC     128
#define OUTC    128
#define ZDIM    512
#define HW      3136          // 56*56
#define HDIM    56
#define FANIN   1152          // 128*3*3
#define NTOT    147456        // OUTC*FANIN
#define WSCALE  0.0416666667f // sqrt(2/1152) = 1/24

#if defined(__gfx1250__) && __has_builtin(__builtin_amdgcn_tensor_load_to_lds)
#define USE_TDM 1
#else
#define USE_TDM 0
#endif

union FragU { v16bf v; uint4 q[2]; };

// ---------------------------------------------------------------------------
// Kernel 0: convert x (f32) -> xh (bf16), one float4 per thread
// ---------------------------------------------------------------------------
__global__ void __launch_bounds__(256) cvt_x(const float* __restrict__ x,
                                             __bf16* __restrict__ xh) {
  int i = (blockIdx.x * 256 + threadIdx.x) * 4;
  float4 v = *(const float4*)(x + i);
  xh[i + 0] = (__bf16)v.x;
  xh[i + 1] = (__bf16)v.y;
  xh[i + 2] = (__bf16)v.z;
  xh[i + 3] = (__bf16)v.w;
}

// ---------------------------------------------------------------------------
// Kernel A: gate g[b][oc] = sigmoid(z[b]·gate_W[:,oc] + gate_b[oc])
// ---------------------------------------------------------------------------
__global__ void __launch_bounds__(256) gate_kernel(const float* __restrict__ z,
                                                   const float* __restrict__ gW,
                                                   const float* __restrict__ gb,
                                                   float* __restrict__ g) {
  int idx = blockIdx.x * 256 + threadIdx.x;   // 0..2047
  int b = idx >> 7, oc = idx & 127;
  float s = gb[oc];
  const float* zr = z + b * ZDIM;
  for (int k = 0; k < ZDIM; ++k) s += zr[k] * gW[k * OUTC + oc];
  g[idx] = 1.f / (1.f + __expf(-s));
}

// ---------------------------------------------------------------------------
// Kernel B: dw[16][147456] = z[16][512] @ dw_W[512][147456] + dw_b   (WMMA)
// One wave computes one 16(batch) x 16(col) tile, K=512 in 16 steps of 32.
// Memory-bound on dw_W (302 MB): ~13 us floor at 23.3 TB/s.
// ---------------------------------------------------------------------------
__global__ void __launch_bounds__(256) hyper_gemm(const float* __restrict__ z,
                                                  const float* __restrict__ dwW,
                                                  const float* __restrict__ dwb,
                                                  float* __restrict__ dw) {
  __shared__ __align__(16) __bf16 zA[16][520];  // padded row stride: bank-spread b128
  int tid = threadIdx.x;
  for (int i = tid; i < 16 * ZDIM; i += 256) {
    int m = i >> 9, k = i & 511;
    zA[m][k] = (__bf16)z[i];                    // coalesced: k fastest
  }
  __syncthreads();

  int wave = tid >> 5, lane = tid & 31;
  int tile = blockIdx.x * 8 + wave;             // 1152 blocks * 8 waves = 9216 tiles
  int n    = tile * 16 + (lane & 15);
  int m    = lane & 15;
  int kA   = (lane < 16) ? 0 : 16;              // A-frag k half
  int kBh  = (lane < 16) ? 0 : 16;              // B-frag k half

  v8f acc = {};
  for (int s = 0; s < 16; ++s) {
    int kb = s * 32;
    FragU a;                                    // A: row m, k = kb+{kA..kA+15}
    a.q[0] = *(const uint4*)&zA[m][kb + kA];
    a.q[1] = *(const uint4*)&zA[m][kb + kA + 8];

    v16bf bfrag;
    const float* col = dwW + (size_t)(kb + kBh) * NTOT + n;
#pragma unroll
    for (int j = 0; j < 16; ++j)                // B: col n, 16 consecutive k rows
      bfrag[j] = (__bf16)col[(size_t)j * NTOT];
    if (s < 15) __builtin_prefetch(col + (size_t)32 * NTOT, 0, 1);

    acc = __builtin_amdgcn_wmma_f32_16x16x32_bf16(false, a.v, false, bfrag,
                                                  (short)0, acc, false, false);
  }
  float bias = dwb[n];
  int mbase = (lane < 16) ? 0 : 8;
#pragma unroll
  for (int i = 0; i < 8; ++i)
    dw[(size_t)(mbase + i) * NTOT + n] = acc[i] + bias;
}

// ---------------------------------------------------------------------------
// Kernel C: per-(b,oc) standardize dw over fan-in, scale by 1/24, fold gate:
//   W_eff = (1-g)*W_base + g*dw_std     -> written as bf16 to wh
// ---------------------------------------------------------------------------
__global__ void __launch_bounds__(256) std_blend(const float* __restrict__ dw,
                                                 const float* __restrict__ basew,
                                                 const float* __restrict__ g,
                                                 __bf16* __restrict__ wh) {
  int grp = blockIdx.x;                         // 0..2047
  int b = grp >> 7, oc = grp & 127;
  const float* w  = dw + (size_t)b * NTOT + (size_t)oc * FANIN;
  const float* bw = basew + (size_t)oc * FANIN;
  __bf16* wo      = wh + (size_t)b * NTOT + (size_t)oc * FANIN;

  __shared__ float ssum[256], ssq[256];
  float s = 0.f, q = 0.f;
  for (int i = threadIdx.x; i < FANIN; i += 256) { float v = w[i]; s += v; q += v * v; }
  ssum[threadIdx.x] = s; ssq[threadIdx.x] = q;
  __syncthreads();
  for (int o = 128; o > 0; o >>= 1) {
    if (threadIdx.x < o) {
      ssum[threadIdx.x] += ssum[threadIdx.x + o];
      ssq[threadIdx.x]  += ssq[threadIdx.x + o];
    }
    __syncthreads();
  }
  float mean = ssum[0] * (1.f / FANIN);
  float var  = ssq[0]  * (1.f / FANIN) - mean * mean;
  float k    = rsqrtf(var + 1e-5f) * WSCALE;
  float gg   = g[grp], og = 1.f - gg;
  for (int i = threadIdx.x; i < FANIN; i += 256)
    wo[i] = (__bf16)(og * bw[i] + gg * ((w[i] - mean) * k));
}

// ---------------------------------------------------------------------------
// Kernel D: per-sample implicit-GEMM 3x3 conv, all-bf16 WMMA.
// grid = (49 pixel-tiles of 64, 16 batches); block = 8 waves (all 128 oc).
// Per K-step: TDM (tensor_load_to_lds) DMAs the 128x32 bf16 weight panel into
// padded LDS (pad 4 DW after every 16 DW -> stride-40 rows) while the block
// builds the 64x32 patch panel; each wave then does 4 WMMAs (16oc x 64px).
// All fragments load as 2x ds_load_b128.
// ---------------------------------------------------------------------------
__global__ void __launch_bounds__(256) conv_wmma(const __bf16* __restrict__ xh,
                                                 const __bf16* __restrict__ wh,
                                                 float* __restrict__ out) {
  __shared__ __align__(16) __bf16 wk[128 * 40]; // A panel [oc][k], stride 40
  __shared__ __align__(16) __bf16 bp[64 * 40];  // B panel [px][k], stride 40
  __shared__ int tab[FANIN];                    // (rel_offset<<4) | kpos
  __shared__ int vmask[64];                     // 9-bit border-validity per px

  int b   = blockIdx.y;
  int p0  = blockIdx.x * 64;
  int tid = threadIdx.x, wave = tid >> 5, lane = tid & 31;

  // k -> (ic,kh,kw) packed table (block-invariant, built once)
  for (int k = tid; k < FANIN; k += 256) {
    int ic = k / 9, r = k - ic * 9, kh = r / 3, kw = r - kh * 3;
    int off = ic * HW + (kh - 1) * HDIM + (kw - 1);
    tab[k] = (off << 4) | r;
  }
  // per-pixel halo validity mask (9 taps)
  if (tid < 64) {
    int p = p0 + tid;
    int h0 = p / HDIM, w0 = p - h0 * HDIM;
    int mask = 0;
#pragma unroll
    for (int pos = 0; pos < 9; ++pos) {
      int dh = pos / 3 - 1, dwp = pos % 3 - 1;
      if ((unsigned)(h0 + dh) < HDIM && (unsigned)(w0 + dwp) < HDIM) mask |= 1 << pos;
    }
    vmask[tid] = mask;
  }
  __syncthreads();

  const __bf16* xb = xh + (size_t)b * INC * HW + p0; // + px + off = patch element

#if USE_TDM
  // Loop-invariant part of the Tensor DMA descriptor (D#):
  // tile 32(k) x 128(oc) of a 1152 x 128 bf16 tensor, row stride 1152,
  // LDS padding: 4 DWORDs after every 16 DWORDs (-> 40-element rows).
  unsigned ldsoff = (unsigned)(size_t)(__bf16*)wk;   // LDS byte offset (addr[31:0])
  unsigned long long wg = (unsigned long long)(size_t)(wh + (size_t)b * NTOT);
  v8i d1 = { (int)((1u << 16) | (1u << 20) | (3u << 22) | (3u << 25)),
             (int)(1152u << 16),          // tensor_dim0 = 1152 (low 16)
             (int)(128u << 16),           // dim0 hi=0 | tensor_dim1 = 128
             (int)(32u << 16),            // dim1 hi=0 | tile_dim0 = 32
             (int)128u,                   // tile_dim1 = 128 | tile_dim2 = 0
             (int)1152u,                  // tensor_dim0_stride = 1152
             0, 0 };
  v4i dz = { 0, 0, 0, 0 };
#else
  const uint32_t* wsrc = (const uint32_t*)(wh + (size_t)b * NTOT);
  uint32_t* wdst = (uint32_t*)wk;
#endif

  int kA  = (lane < 16) ? 0 : 16;
  int kBh = (lane < 16) ? 0 : 16;
  int m   = lane & 15;
  int ocb = wave * 16;

  v8f acc[4] = {{}, {}, {}, {}};
  for (int s = 0; s < 36; ++s) {
    int kb = s * 32;
#if USE_TDM
    if (wave == 0) {                             // one wave issues the DMA
      unsigned long long ga = wg + (unsigned long long)kb * 2;
      v4u d0 = { 1u,                             // count=1, user descriptor
                 ldsoff,
                 (unsigned)ga,
                 (unsigned)(ga >> 32) | (2u << 30) };  // addr[56:32] | type=2
#if __clang_major__ >= 23
      __builtin_amdgcn_tensor_load_to_lds(d0, d1, dz, dz, (v8i){0,0,0,0,0,0,0,0}, 0);
#else
      __builtin_amdgcn_tensor_load_to_lds(d0, d1, dz, dz, 0);
#endif
      __builtin_amdgcn_s_wait_tensorcnt(0);
    }
#else
    // stage A panel: 128 oc x 32 k bf16, copied as u32 pairs (coalesced)
    for (int i = tid; i < 128 * 16; i += 256) {
      int oc = i >> 4, kp = i & 15;
      wdst[oc * 20 + kp] = wsrc[oc * 576 + (kb >> 1) + kp];
    }
#endif
    // build B panel: 64 px x 32 k, gathered once for all 8 waves
    for (int i = tid; i < 64 * 32; i += 256) {
      int px = i >> 5, k = i & 31;
      int t   = tab[kb + k];
      int pos = t & 15;
      int off = t >> 4;                        // arithmetic shift (off may be <0)
      int valid = (vmask[px] >> pos) & 1;
      int idx = valid ? (px + off) : 0;        // clamped: always in-bounds
      __bf16 v = xb[idx];
      bp[px * 40 + k] = valid ? v : (__bf16)0.f;
    }
    __syncthreads();

    FragU a;
    a.q[0] = *(const uint4*)&wk[(ocb + m) * 40 + kA];
    a.q[1] = *(const uint4*)&wk[(ocb + m) * 40 + kA + 8];
#pragma unroll
    for (int t = 0; t < 4; ++t) {
      FragU bf;
      int px = t * 16 + (lane & 15);
      bf.q[0] = *(const uint4*)&bp[px * 40 + kBh];
      bf.q[1] = *(const uint4*)&bp[px * 40 + kBh + 8];
      acc[t] = __builtin_amdgcn_wmma_f32_16x16x32_bf16(false, a.v, false, bf.v,
                                                       (short)0, acc[t], false, false);
    }
    __syncthreads();
  }

  int mbase = (lane < 16) ? 0 : 8;
  float* ob = out + (size_t)b * OUTC * HW + p0;
#pragma unroll
  for (int t = 0; t < 4; ++t)
#pragma unroll
    for (int i = 0; i < 8; ++i)
      ob[(size_t)(ocb + mbase + i) * HW + t * 16 + (lane & 15)] = acc[t][i];
}

// ---------------------------------------------------------------------------
extern "C" void kernel_launch(void* const* d_in, const int* in_sizes, int n_in,
                              void* d_out, int out_size, void* d_ws, size_t ws_size,
                              hipStream_t stream) {
  const float* x   = (const float*)d_in[0];
  const float* z   = (const float*)d_in[1];
  const float* bw  = (const float*)d_in[2];
  const float* dwW = (const float*)d_in[3];
  const float* dwb = (const float*)d_in[4];
  const float* gW  = (const float*)d_in[5];
  const float* gb  = (const float*)d_in[6];
  float* out = (float*)d_out;

  // workspace layout (~27 MB)
  float*  dw = (float*)d_ws;                       // [16][147456] f32
  float*  g  = dw + (size_t)BATCH * NTOT;          // [16][128]   f32
  __bf16* wh = (__bf16*)(g + 2048);                // [16][147456] bf16
  __bf16* xh = wh + (size_t)BATCH * NTOT;          // [16][128][3136] bf16

  cvt_x      <<<6272, 256, 0, stream>>>(x, xh);    // 16*128*3136/4 threads
  gate_kernel<<<8,    256, 0, stream>>>(z, gW, gb, g);
  hyper_gemm <<<1152, 256, 0, stream>>>(z, dwW, dwb, dw);
  std_blend  <<<2048, 256, 0, stream>>>(dw, bw, g, wh);
  conv_wmma  <<<dim3(49, 16), 256, 0, stream>>>(xh, wh, out);
}
